// ProteinLigandGNN_54193897341260
// MI455X (gfx1250) — matrix operand order repaired and checked by reading, not compiled, not run on hardware
//
#include <hip/hip_runtime.h>
#include <hip/hip_bf16.h>

// ---------------------------------------------------------------------------
// ProteinLigandGNN for MI455X (gfx1250, wave32, WMMA + TDM).
//
//   per tower (protein, ligand):
//     dinv  = rsqrt(1 + in_degree)                     (atomic histogram)
//     xw    = h @ W       (WMMA f16-in/f32-acc GEMM; A/B tiles staged into
//                          LDS by the Tensor Data Mover, which also does the
//                          K zero-padding via tensor_dim OOB=0 semantics)
//     agg   = scatter_add(edges, xw[src]*dinv[src]*dinv[dst])  (f32 atomics,
//             coalesced float4 per wave, prefetch of next source row)
//     h     = relu(agg + dinv^2*xw + b)                (self-loop fused)
//     pool -> combined[:, tower]
//   hidden = relu(combined @ Wfc1 + bfc1)              (WMMA GEMM)
//   out    = three 128->1 heads (wave-shuffle dot products)
// ---------------------------------------------------------------------------

typedef __attribute__((ext_vector_type(16))) _Float16     v16h;
typedef __attribute__((ext_vector_type(8)))  float        v8f;
typedef __attribute__((ext_vector_type(4)))  unsigned int u32x4;
typedef __attribute__((ext_vector_type(8)))  int          i32x8;
typedef __attribute__((ext_vector_type(4)))  int          i32x4;

#define HDIM   128
#define NGRAPH 256

// ---------------------------------------------------------------- TDM helper
// 2D tile load Global->LDS via Tensor Data Mover.
//   dim0/dim1: remaining tensor extent (elements); OOB reads return 0
//   stride0  : row stride in elements; tile0 x tile1: tile shape
//   padbits  : group1 word0 bits 20..31 (pad_enable/interval/amount)
// Issue from ONE wave per block; follow with s_wait_tensorcnt + barrier.
__device__ __forceinline__ void tdm_load_2d(unsigned lds_off, const void* gptr,
                                            unsigned dim0, unsigned dim1,
                                            unsigned long long stride0,
                                            unsigned tile0, unsigned tile1,
                                            unsigned padbits) {
  unsigned long long ga = (unsigned long long)gptr;
  u32x4 g0;
  g0[0] = 1u;                                   // count=1, user descriptor
  g0[1] = lds_off;                              // LDS byte address
  g0[2] = (unsigned)ga;                         // global_addr[31:0]
  g0[3] = (unsigned)((ga >> 32) & 0x1FFFFFFu)   // global_addr[56:32]
          | 0x80000000u;                        // type=2 ("image")
  i32x8 g1;
  g1[0] = (int)((2u << 16) | padbits);          // data_size=4B + pad cfg
  g1[1] = (int)((dim0 & 0xFFFFu) << 16);        // tensor_dim0[15:0]
  g1[2] = (int)(((dim0 >> 16) & 0xFFFFu) | ((dim1 & 0xFFFFu) << 16));
  g1[3] = (int)(((dim1 >> 16) & 0xFFFFu) | ((tile0 & 0xFFFFu) << 16));
  g1[4] = (int)(tile1 & 0xFFFFu);               // tile_dim1 (tile_dim2=0 -> 2D)
  g1[5] = (int)(unsigned)stride0;               // tensor_dim0_stride[31:0]
  g1[6] = (int)((stride0 >> 32) & 0xFFFFu);     // stride[47:32]; dim1_stride=0
  g1[7] = 0;
  i32x4 z4 = {0, 0, 0, 0};
#if __clang_major__ >= 23
  i32x8 z8 = {0, 0, 0, 0, 0, 0, 0, 0};
  __builtin_amdgcn_tensor_load_to_lds(g0, g1, z4, z4, z8, 0);
#else
  __builtin_amdgcn_tensor_load_to_lds(g0, g1, z4, z4, 0);
#endif
}

__device__ __forceinline__ unsigned lds_addr_of(const void* p) {
  // Flat LDS-aperture address: addr[31:0] is the LDS byte offset.
  return (unsigned)(unsigned long long)p;
}

// ---------------------------------------------------------------- utilities
__global__ void k_fill(float* __restrict__ p, long n, float v) {
  long i = (long)blockIdx.x * blockDim.x + threadIdx.x;
  long stride = (long)gridDim.x * blockDim.x;
  for (; i < n; i += stride) p[i] = v;
}

__global__ void k_deg_accum(const int* __restrict__ dst, int E, float* __restrict__ deg) {
  int i = blockIdx.x * blockDim.x + threadIdx.x;
  int stride = gridDim.x * blockDim.x;
  for (; i < E; i += stride) atomicAdd(&deg[dst[i]], 1.0f);
}

__global__ void k_rsqrt_inplace(float* __restrict__ d, int n) {
  int i = blockIdx.x * blockDim.x + threadIdx.x;
  int stride = gridDim.x * blockDim.x;
  for (; i < n; i += stride) {
    float v = d[i];
    d[i] = (v > 0.0f) ? rsqrtf(v) : 0.0f;
  }
}

// ------------------------------------------------------------- WMMA GEMM
// C[M x N] = A[M x K] @ B[K x N], row-major f32, N == 128, M % 16 == 0.
// Block = 256 threads = 8 waves: block owns 16 rows, wave w owns cols
// [16w, 16w+16). Per 32-wide K step the TDM stages:
//   A tile : 16 x 32 f32, row stride K, LDS-padded to 33-dword rows
//            (conflict-free A-fragment reads), K-overrun zero-filled by TDM
//   B panel: 32 x 128 f32, packed
// Fragments built from LDS with v_cvt_f16_f32 (documented wave32 layouts):
//   A (16x32 f16): lane {khi=lane>>4, mrow=lane&15};
//                  VGPR j holds K = (j<4?0:16) + (j&3)*2 + khi*8 + {0,1}
//   B (32x16)    : lane holds col N = lane&15; half-slot s: K = (s<8?0:16) + khi*8 + (s&7)
//   C/D (f32)    : VGPR r holds row r + khi*8, col lane&15
__global__ void __launch_bounds__(256)
k_gemm_wmma(const float* __restrict__ A, const float* __restrict__ B,
            float* __restrict__ C, int M, int K, int N) {
  __shared__ float A_lds[16 * 33];       // 33-dword rows via TDM padding
  __shared__ float B_lds[32 * 128];

  const int wave = threadIdx.x >> 5;
  const int lane = threadIdx.x & 31;
  const int mrow = lane & 15;
  const int khi  = lane >> 4;
  const int row0 = blockIdx.x * 16;
  const int col0 = wave * 16;

  const unsigned a_off = lds_addr_of(A_lds);
  const unsigned b_off = lds_addr_of(B_lds);
  // pad_enable | pad_interval=4 (pad after every 32 dwords) | pad_amount=0 (1 dword)
  const unsigned apad = (1u << 20) | (4u << 22);

  v8f acc = {};
  for (int kb = 0; kb < K; kb += 32) {
    if (threadIdx.x < 32) {              // one wave issues both DMAs
      tdm_load_2d(a_off, A + (size_t)row0 * K + kb,
                  (unsigned)(K - kb), 16u, (unsigned long long)K, 32u, 16u, apad);
      tdm_load_2d(b_off, B + (size_t)kb * N,
                  (unsigned)N, (unsigned)(K - kb), (unsigned long long)N, 128u, 32u, 0u);
      __builtin_amdgcn_s_wait_tensorcnt(0);
    }
    __syncthreads();

    v16h a, b;
#pragma unroll
    for (int j = 0; j < 8; ++j) {
      const int kbase = ((j < 4) ? 0 : 16) + (j & 3) * 2 + khi * 8;
      const int idx = mrow * 33 + kbase;
      a[2 * j + 0] = (_Float16)A_lds[idx + 0];
      a[2 * j + 1] = (_Float16)A_lds[idx + 1];
    }
#pragma unroll
    for (int s = 0; s < 16; ++s) {
      b[s] = (_Float16)B_lds[(((s & 8) << 1) + khi * 8 + (s & 7)) * 128 + col0 + mrow];
    }
    acc = __builtin_amdgcn_wmma_f32_16x16x32_f16(
        /*neg_a=*/false, a, /*neg_b=*/false, b,
        /*c_mod=*/(short)0, acc, /*reuse_a=*/false, /*reuse_b=*/false);
    __syncthreads();                     // protect LDS before next DMA
  }
#pragma unroll
  for (int r = 0; r < 8; ++r) {
    const int rr = row0 + r + khi * 8;
    C[(size_t)rr * N + col0 + mrow] = acc[r];
  }
}

// ---------------------------------------------------- edge scatter-gather
// One wave per edge (grid-stride). 128-float rows = 32 lanes x float4.
// Prefetch the next edge's source row (random access, latency bound).
__global__ void k_edge_agg(const int* __restrict__ src, const int* __restrict__ dst,
                           int E, const float* __restrict__ xw,
                           const float* __restrict__ dinv, float* __restrict__ agg) {
  const int wpb = blockDim.x >> 5;
  const int lane = threadIdx.x & 31;
  int e = blockIdx.x * wpb + (threadIdx.x >> 5);
  const int stride = gridDim.x * wpb;
  for (; e < E; e += stride) {
    const int s = src[e];
    const int d = dst[e];
    const int en = e + stride;
    if (en < E) {
      const int sn = src[en];
      __builtin_prefetch(xw + (size_t)sn * HDIM + lane * 4, 0, 0);  // global_prefetch_b8
    }
    const float nrm = dinv[s] * dinv[d];
    const float4 v = ((const float4*)(xw + (size_t)s * HDIM))[lane];
    float* arow = agg + (size_t)d * HDIM + lane * 4;
    atomicAdd(arow + 0, v.x * nrm);
    atomicAdd(arow + 1, v.y * nrm);
    atomicAdd(arow + 2, v.z * nrm);
    atomicAdd(arow + 3, v.w * nrm);
  }
}

// h = relu(agg + dinv^2 * xw + bias), in place on agg. (self-loop term fused)
__global__ void k_finalize(float* __restrict__ agg, const float* __restrict__ xw,
                           const float* __restrict__ dinv, const float* __restrict__ bias,
                           long total /* n * HDIM */) {
  long i = (long)blockIdx.x * blockDim.x + threadIdx.x;
  long stride = (long)gridDim.x * blockDim.x;
  for (; i < total; i += stride) {
    const long node = i >> 7;
    const int c = (int)(i & (HDIM - 1));
    const float di = dinv[node];
    const float v = agg[i] + di * di * xw[i] + bias[c];
    agg[i] = fmaxf(v, 0.0f);
  }
}

// ------------------------------------------------------------- mean pool
__global__ void k_pool_accum(const float* __restrict__ h, const int* __restrict__ batch,
                             long n, float* __restrict__ pool, int stride_cols, int colofs) {
  long i = (long)blockIdx.x * blockDim.x + threadIdx.x;
  long stride = (long)gridDim.x * blockDim.x;
  const long total = n * HDIM;
  for (; i < total; i += stride) {
    const long node = i >> 7;
    const int c = (int)(i & (HDIM - 1));
    const int g = batch[node];
    atomicAdd(&pool[(size_t)g * stride_cols + colofs + c], h[i]);
  }
}

__global__ void k_count_accum(const int* __restrict__ batch, int n, float* __restrict__ cnt) {
  int i = blockIdx.x * blockDim.x + threadIdx.x;
  int stride = gridDim.x * blockDim.x;
  for (; i < n; i += stride) atomicAdd(&cnt[batch[i]], 1.0f);
}

__global__ void k_pool_div(float* __restrict__ pool, const float* __restrict__ cnt,
                           int stride_cols, int colofs) {
  int i = blockIdx.x * blockDim.x + threadIdx.x;          // over NGRAPH*HDIM
  const int g = i >> 7;
  const int c = i & (HDIM - 1);
  if (g < NGRAPH) pool[(size_t)g * stride_cols + colofs + c] /= fmaxf(cnt[g], 1.0f);
}

__global__ void k_bias_relu(float* __restrict__ x, const float* __restrict__ bias, long total) {
  long i = (long)blockIdx.x * blockDim.x + threadIdx.x;
  long stride = (long)gridDim.x * blockDim.x;
  for (; i < total; i += stride) {
    const int c = (int)(i & (HDIM - 1));
    x[i] = fmaxf(x[i] + bias[c], 0.0f);
  }
}

// ------------------------------------------------------- 128->1 heads (x3)
__global__ void k_heads(const float* __restrict__ hid,
                        const float* __restrict__ w0, const float* __restrict__ b0,
                        const float* __restrict__ w1, const float* __restrict__ b1,
                        const float* __restrict__ w2, const float* __restrict__ b2,
                        float* __restrict__ out, int G) {
  const int wave = threadIdx.x >> 5;
  const int lane = threadIdx.x & 31;
  const int g = blockIdx.x * (blockDim.x >> 5) + wave;
  if (g >= G) return;
  const float4 v = ((const float4*)(hid + (size_t)g * HDIM))[lane];
  const float* ws[3] = {w0, w1, w2};
  const float* bs[3] = {b0, b1, b2};
#pragma unroll
  for (int hd = 0; hd < 3; ++hd) {
    const float4 w = ((const float4*)ws[hd])[lane];
    float p = v.x * w.x + v.y * w.y + v.z * w.z + v.w * w.w;
    for (int off = 16; off > 0; off >>= 1) p += __shfl_down(p, off);
    if (lane == 0) out[hd * G + g] = p + bs[hd][0];
  }
}

// ---------------------------------------------------------------------------
extern "C" void kernel_launch(void* const* d_in, const int* in_sizes, int n_in,
                              void* d_out, int out_size, void* d_ws, size_t ws_size,
                              hipStream_t stream) {
  const long NP = 100000, NL = 8192;
  const int  EP = 1600000, EL = 65536;

  const float* protein_x = (const float*)d_in[0];
  const int*   p_ei      = (const int*)d_in[1];
  const int*   p_batch   = (const int*)d_in[2];
  const float* ligand_x  = (const float*)d_in[3];
  const int*   l_ei      = (const int*)d_in[4];
  const int*   l_batch   = (const int*)d_in[5];
  // d_in[6] = num_graphs (== NGRAPH)
  const float* Wp1 = (const float*)d_in[7];   const float* bp1 = (const float*)d_in[8];
  const float* Wp2 = (const float*)d_in[9];   const float* bp2 = (const float*)d_in[10];
  const float* Wl1 = (const float*)d_in[11];  const float* bl1 = (const float*)d_in[12];
  const float* Wl2 = (const float*)d_in[13];  const float* bl2 = (const float*)d_in[14];
  const float* Wfc1= (const float*)d_in[15];  const float* bfc1= (const float*)d_in[16];
  const float* Wpkd= (const float*)d_in[17];  const float* bpkd= (const float*)d_in[18];
  const float* Wpki= (const float*)d_in[19];  const float* bpki= (const float*)d_in[20];
  const float* Wba = (const float*)d_in[21];  const float* bba = (const float*)d_in[22];
  float* out = (float*)d_out;

  // ---- workspace carve-up (floats), ~111 MB total --------------------
  float* ws = (float*)d_ws;
  float* A_p    = ws;                 // NP*H  (xw ping)
  float* B_p    = A_p + NP * HDIM;    // NP*H  (agg / h pong)
  float* A_l    = B_p + NP * HDIM;    // NL*H
  float* B_l    = A_l + NL * HDIM;    // NL*H
  float* dinv_p = B_l + NL * HDIM;    // NP
  float* dinv_l = dinv_p + NP;        // NL
  float* cnt_p  = dinv_l + NL;        // NGRAPH
  float* cnt_l  = cnt_p + NGRAPH;     // NGRAPH
  float* combined = cnt_l + NGRAPH;   // NGRAPH * 2H
  float* hidden   = combined + NGRAPH * 2 * HDIM;  // NGRAPH * H

  const int* p_src = p_ei;      const int* p_dst = p_ei + EP;
  const int* l_src = l_ei;      const int* l_dst = l_ei + EL;

  const int T = 256;
  auto blocks = [](long n, int t) { return dim3((unsigned)((n + t - 1) / t)); };

  // ---- degrees -> dinv ----------------------------------------------
  k_fill<<<blocks(NP, T), T, 0, stream>>>(dinv_p, NP, 1.0f);   // self loop
  k_fill<<<blocks(NL, T), T, 0, stream>>>(dinv_l, NL, 1.0f);
  k_deg_accum<<<2048, T, 0, stream>>>(p_dst, EP, dinv_p);
  k_deg_accum<<<256,  T, 0, stream>>>(l_dst, EL, dinv_l);
  k_rsqrt_inplace<<<blocks(NP, T), T, 0, stream>>>(dinv_p, (int)NP);
  k_rsqrt_inplace<<<blocks(NL, T), T, 0, stream>>>(dinv_l, (int)NL);

  // zero pooling accumulators once
  k_fill<<<blocks(NGRAPH * 2 * HDIM, T), T, 0, stream>>>(combined, NGRAPH * 2 * HDIM, 0.0f);
  k_fill<<<1, T, 0, stream>>>(cnt_p, NGRAPH, 0.0f);
  k_fill<<<1, T, 0, stream>>>(cnt_l, NGRAPH, 0.0f);

  // ---- protein tower -------------------------------------------------
  k_gemm_wmma<<<dim3((unsigned)(NP / 16)), T, 0, stream>>>(protein_x, Wp1, A_p, (int)NP, 23, HDIM);
  k_fill<<<4096, T, 0, stream>>>(B_p, NP * HDIM, 0.0f);
  k_edge_agg<<<4096, T, 0, stream>>>(p_src, p_dst, EP, A_p, dinv_p, B_p);
  k_finalize<<<4096, T, 0, stream>>>(B_p, A_p, dinv_p, bp1, NP * HDIM);
  k_gemm_wmma<<<dim3((unsigned)(NP / 16)), T, 0, stream>>>(B_p, Wp2, A_p, (int)NP, HDIM, HDIM);
  k_fill<<<4096, T, 0, stream>>>(B_p, NP * HDIM, 0.0f);
  k_edge_agg<<<4096, T, 0, stream>>>(p_src, p_dst, EP, A_p, dinv_p, B_p);
  k_finalize<<<4096, T, 0, stream>>>(B_p, A_p, dinv_p, bp2, NP * HDIM);
  k_pool_accum<<<4096, T, 0, stream>>>(B_p, p_batch, NP, combined, 2 * HDIM, 0);
  k_count_accum<<<blocks(NP, T), T, 0, stream>>>(p_batch, (int)NP, cnt_p);
  k_pool_div<<<blocks(NGRAPH * HDIM, T), T, 0, stream>>>(combined, cnt_p, 2 * HDIM, 0);

  // ---- ligand tower --------------------------------------------------
  k_gemm_wmma<<<dim3((unsigned)(NL / 16)), T, 0, stream>>>(ligand_x, Wl1, A_l, (int)NL, 4, HDIM);
  k_fill<<<1024, T, 0, stream>>>(B_l, NL * HDIM, 0.0f);
  k_edge_agg<<<1024, T, 0, stream>>>(l_src, l_dst, EL, A_l, dinv_l, B_l);
  k_finalize<<<1024, T, 0, stream>>>(B_l, A_l, dinv_l, bl1, NL * HDIM);
  k_gemm_wmma<<<dim3((unsigned)(NL / 16)), T, 0, stream>>>(B_l, Wl2, A_l, (int)NL, HDIM, HDIM);
  k_fill<<<1024, T, 0, stream>>>(B_l, NL * HDIM, 0.0f);
  k_edge_agg<<<1024, T, 0, stream>>>(l_src, l_dst, EL, A_l, dinv_l, B_l);
  k_finalize<<<1024, T, 0, stream>>>(B_l, A_l, dinv_l, bl2, NL * HDIM);
  k_pool_accum<<<1024, T, 0, stream>>>(B_l, l_batch, NL, combined, 2 * HDIM, HDIM);
  k_count_accum<<<blocks(NL, T), T, 0, stream>>>(l_batch, (int)NL, cnt_l);
  k_pool_div<<<blocks(NGRAPH * HDIM, T), T, 0, stream>>>(combined, cnt_l, 2 * HDIM, HDIM);

  // ---- MLP head ------------------------------------------------------
  k_gemm_wmma<<<dim3(NGRAPH / 16), T, 0, stream>>>(combined, Wfc1, hidden, NGRAPH, 2 * HDIM, HDIM);
  k_bias_relu<<<blocks((long)NGRAPH * HDIM, T), T, 0, stream>>>(hidden, bfc1, (long)NGRAPH * HDIM);
  k_heads<<<dim3(NGRAPH / 8), T, 0, stream>>>(hidden, Wpkd, bpkd, Wpki, bpki, Wba, bba, out, NGRAPH);
}
